// scattering_2d_2layer_partial_cov_21174188769980
// MI455X (gfx1250) — compile-verified, matrix operand order checked by reading, not run on hardware
//
#include <hip/hip_runtime.h>
#include <hip/hip_bf16.h>
#include <math.h>

typedef __attribute__((ext_vector_type(16))) __bf16 v16bf;
typedef __attribute__((ext_vector_type(8)))  float  v8f;

#define NSIDE 256
#define NN    65536
#define F1    81
#define F2    41
#define F2P   48
#define NPAIR 1521
#define NPAIR_PAD 1536
#define NTILE 96
#define KSEG  8
#define NCHUNK 2048
#define EPS_K 1e-8f

// ---------------- prep: conv_hat = x_c * psi (complex mult), 81 channels ----
__global__ void k_convhat(const float* __restrict__ xh,
                          const float* __restrict__ pr,
                          const float* __restrict__ pi_,
                          float2* __restrict__ out) {
  int idx = blockIdx.x * 256 + threadIdx.x;         // < F1*NN
  int xy = idx & (NN - 1);
  float xr = xh[2 * xy], xi = xh[2 * xy + 1];
  float a = pr[idx], b = pi_[idx];
  out[idx] = make_float2(xr * a - xi * b, xr * b + xi * a);
}

// -- prep: psi2 = r^2+i^2 -> bf16, pre-swizzled into WMMA B-fragment order --
// Bsw[chunk][ftile][lane][e], 32B per lane, so the GEMM loads fragments
// directly from global (2x global_load_b128, fully coalesced, L2-resident).
__global__ void k_psi2sw(const float* __restrict__ r2,
                         const float* __restrict__ i2,
                         __bf16* __restrict__ out) {
  int id = blockIdx.x * 256 + threadIdx.x;          // < NCHUNK*3*32
  int lane  = id & 31;
  int ftile = (id >> 5) % 3;
  int chunk = id / 96;
  int f  = ftile * 16 + (lane & 15);
  int kh = lane >> 4;
  int pixBase = chunk << 5;
#pragma unroll
  for (int e = 0; e < 16; ++e) {
    int k = ((e & 8) << 1) + (kh << 3) + (e & 7);
    float v = 0.f;
    if (f < F2) {
      size_t s = (size_t)f * NN + pixBase + k;
      float a = r2[s], b = i2[s];
      v = a * a + b * b;
    }
    out[(size_t)id * 16 + e] = (__bf16)v;
  }
}

// ---------------- 256-point radix-2 FFT along rows or cols, in LDS ---------
__global__ void fft1d(float2* __restrict__ buf, int cols, float dsign, float scale) {
  __shared__ float2 lds[256];
  int line = blockIdx.x;
  int ch = line >> 8, idx = line & 255;
  size_t base; int stride;
  if (cols) { base = (size_t)ch * NN + idx;                 stride = NSIDE; }
  else      { base = (size_t)ch * NN + (size_t)idx * NSIDE; stride = 1; }
  int t = threadIdx.x;                                // 0..127
  for (int i = t; i < 256; i += 128) {
    int r = (int)(__brev((unsigned)i) >> 24);
    lds[r] = buf[base + (size_t)i * stride];
  }
  __syncthreads();
  for (int s = 1; s <= 8; ++s) {
    int half = 1 << (s - 1);
    int grp = t >> (s - 1);
    int pos = t & (half - 1);
    int i0 = (grp << s) + pos;
    int i1 = i0 + half;
    float ang = dsign * 6.28318530717958647f * (float)pos / (float)(1 << s);
    float sw, cw; __sincosf(ang, &sw, &cw);
    float2 u = lds[i0], v = lds[i1];
    float2 vw = make_float2(v.x * cw - v.y * sw, v.x * sw + v.y * cw);
    lds[i0] = make_float2(u.x + vw.x, u.y + vw.y);
    lds[i1] = make_float2(u.x - vw.x, u.y - vw.y);
    __syncthreads();
  }
  for (int i = t; i < 256; i += 128) {
    float2 c = lds[i];
    buf[base + (size_t)i * stride] = make_float2(c.x * scale, c.y * scale);
  }
}

// ---------------- modulus (in place, imag -> 0, ready for fft2) ------------
__global__ void k_mod(float2* __restrict__ buf) {
  int idx = blockIdx.x * 256 + threadIdx.x;           // < F1*NN
  float2 c = buf[idx];
  buf[idx] = make_float2(sqrtf(c.x * c.x + c.y * c.y + EPS_K), 0.f);
}

// ---------------- fused cross-spectrum x psi2 GEMM via WMMA bf16 -----------
// Res[pair,f] = sum_xy P[r1]*conj(P[r2])*psi2[f]; 16 pairs/block, K split in 8
__global__ __launch_bounds__(256) void k_gemm(const float2* __restrict__ P,
                                              const v16bf* __restrict__ Bsw,
                                              float* __restrict__ partial) {
  __shared__ char smemRaw[49152];
  __shared__ int rows[16][2];
  const int tid = threadIdx.x;
  const int w = tid >> 5, t = tid & 31;
  const int pairBase = blockIdx.x * 16;
  const int seg = blockIdx.y;

  if (tid < 16) {
    int p = pairBase + tid, r1, r2;
    if (p < 640)       { int a = p / 80, b = (p / 10) % 8, j = p % 10; r1 = a * 10 + j; r2 = b * 10 + j; }
    else if (p < 1440) { int q = p - 640; int i = q / 100, j = (q / 10) % 10, l = q % 10; r1 = i * 10 + j; r2 = i * 10 + l; }
    else if (p < 1520) { int q = p - 1440; int i = q / 10, j = q % 10; r1 = i * 10 + j; r2 = 80; }
    else               { r1 = 80; r2 = 80; }
    rows[tid][0] = r1; rows[tid][1] = r2;
  }
  __syncthreads();

  __bf16* Are = (__bf16*)smemRaw + w * 512;           // [16][32] per wave
  __bf16* Aim = (__bf16*)smemRaw + 4096 + w * 512;    // [16][32] per wave

  v8f accRe0 = {}, accRe1 = {}, accRe2 = {};
  v8f accIm0 = {}, accIm1 = {}, accIm2 = {};
  const int mrow = t & 15, kh = t >> 4;

  for (int it = 0; it < 32; ++it) {
    int chunk = seg * 256 + w + (it << 3);
    int pix = (chunk << 5) + t;
    // build cross-spectrum A tiles (per-wave LDS, same-wave DS is in-order)
#pragma unroll
    for (int m = 0; m < 16; ++m) {
      float2 p1 = P[rows[m][0] * NN + pix];
      float2 p2 = P[rows[m][1] * NN + pix];
      Are[m * 32 + t] = (__bf16)(p1.x * p2.x + p1.y * p2.y);
      Aim[m * 32 + t] = (__bf16)(p1.y * p2.x - p1.x * p2.y);
    }
    // B fragments direct from pre-swizzled global (coalesced 32B per lane)
    v16bf b0 = Bsw[(chunk * 3 + 0) * 32 + t];
    v16bf b1 = Bsw[(chunk * 3 + 1) * 32 + t];
    v16bf b2 = Bsw[(chunk * 3 + 2) * 32 + t];
    // A fragments per CDNA5 16-bit 16x32 layout
    v16bf aRe, aIm;
#pragma unroll
    for (int e = 0; e < 16; ++e) {
      int k = ((e & 8) << 1) + (kh << 3) + (e & 7);
      aRe[e] = Are[mrow * 32 + k];
      aIm[e] = Aim[mrow * 32 + k];
    }
    accRe0 = __builtin_amdgcn_wmma_f32_16x16x32_bf16(false, aRe, false, b0, (short)0, accRe0, false, false);
    accRe1 = __builtin_amdgcn_wmma_f32_16x16x32_bf16(false, aRe, false, b1, (short)0, accRe1, false, false);
    accRe2 = __builtin_amdgcn_wmma_f32_16x16x32_bf16(false, aRe, false, b2, (short)0, accRe2, false, false);
    accIm0 = __builtin_amdgcn_wmma_f32_16x16x32_bf16(false, aIm, false, b0, (short)0, accIm0, false, false);
    accIm1 = __builtin_amdgcn_wmma_f32_16x16x32_bf16(false, aIm, false, b1, (short)0, accIm1, false, false);
    accIm2 = __builtin_amdgcn_wmma_f32_16x16x32_bf16(false, aIm, false, b2, (short)0, accIm2, false, false);
  }

  __syncthreads();
  float* red = (float*)smemRaw;                        // [6][8 waves][256]
#pragma unroll
  for (int v = 0; v < 8; ++v) {
    red[((0 * 8 + w) << 8) + (t << 3) + v] = accRe0[v];
    red[((1 * 8 + w) << 8) + (t << 3) + v] = accRe1[v];
    red[((2 * 8 + w) << 8) + (t << 3) + v] = accRe2[v];
    red[((3 * 8 + w) << 8) + (t << 3) + v] = accIm0[v];
    red[((4 * 8 + w) << 8) + (t << 3) + v] = accIm1[v];
    red[((5 * 8 + w) << 8) + (t << 3) + v] = accIm2[v];
  }
  __syncthreads();
  for (int vi = tid; vi < 768; vi += 256) {
    int ft = vi >> 8, slot = vi & 255;
    float sr = 0.f, si = 0.f;
#pragma unroll
    for (int ww = 0; ww < 8; ++ww) {
      sr += red[((ft * 8 + ww) << 8) + slot];
      si += red[(((ft + 3) * 8 + ww) << 8) + slot];
    }
    int lane = slot >> 3, v = slot & 7;
    int M = v + ((lane >> 4) << 3);
    int fcol = ft * 16 + (lane & 15);
    int pair = pairBase + M;
    size_t o = (((size_t)seg * NPAIR_PAD + pair) * F2P + fcol) * 2;
    partial[o] = sr; partial[o + 1] = si;
  }
}

// ---------------- sum K segments, scale by 1/N^2, magnitude ----------------
__global__ void k_finalize(const float* __restrict__ partial, float* __restrict__ mags) {
  int idx = blockIdx.x * 256 + threadIdx.x;            // < NPAIR_PAD*F2P
  float sr = 0.f, si = 0.f;
  for (int s = 0; s < KSEG; ++s) {
    size_t o = ((size_t)s * NPAIR_PAD * F2P + idx) * 2;
    sr += partial[o]; si += partial[o + 1];
  }
  const float nrm = 1.0f / 65536.0f;
  sr *= nrm; si *= nrm;
  mags[idx] = sqrtf(sr * sr + si * si + EPS_K);
}

// ---------------- build output ordering map (reference loop nest) ----------
__global__ void k_buildmap(int2* __restrict__ map) {
  int pos = 0;
  map[pos++] = make_int2(1520, 40);                    // s0 = (a_hat,a_hat)@f=40
  for (int j = 0; j < 10; ++j) {
    int fj = j / 2; if (fj > 4) fj = 4; int fs = fj * 8;
    for (int i = 0; i < 8; ++i) {
      for (int f = fs; f < 41; ++f)
        for (int b = i; b < 8; ++b)
          map[pos++] = make_int2(i * 80 + b * 10 + j, f);     // Mmag[f,i,b,j]
      for (int l = j + 1; l < 10; ++l) {
        int fl = l / 2; if (fl > 4) fl = 4; int fs2 = fl * 8;
        for (int f = fs2; f < 41; ++f)
          map[pos++] = make_int2(640 + i * 100 + j * 10 + l, f); // Nmag[f,i,j,l]
      }
      for (int f = 0; f < 9; ++f)
        map[pos++] = make_int2(1440 + i * 10 + j, 32 + f);    // Amag[f,i,j]
    }
  }
}

__global__ void k_gather(const int2* __restrict__ map, const float* __restrict__ mags,
                         float* __restrict__ out, int total) {
  int i = blockIdx.x * 256 + threadIdx.x;
  if (i < total) {
    int2 m = map[i];
    out[i] = mags[m.x * F2P + m.y];
  }
}

extern "C" void kernel_launch(void* const* d_in, const int* in_sizes, int n_in,
                              void* d_out, int out_size, void* d_ws, size_t ws_size,
                              hipStream_t stream) {
  const float* x_hat  = (const float*)d_in[0];
  const float* psi_r  = (const float*)d_in[1];
  const float* psi_i  = (const float*)d_in[2];
  const float* psi_r2 = (const float*)d_in[3];
  const float* psi_i2 = (const float*)d_in[4];

  char* ws = (char*)d_ws;
  size_t off = 0;
  float2*  cplx    = (float2*)(ws + off); off += (size_t)F1 * NN * sizeof(float2);        // 42.5 MB
  __bf16*  psi2sw  = (__bf16*)(ws + off); off += (size_t)F2P * NN * sizeof(__bf16);       //  6.3 MB (swizzled)
  float*   partial = (float*)(ws + off);  off += (size_t)KSEG * NPAIR_PAD * F2P * 2 * sizeof(float); // 4.7 MB
  float*   mags    = (float*)(ws + off);  off += (size_t)NPAIR_PAD * F2P * sizeof(float); // 0.3 MB
  int2*    map     = (int2*)(ws + off);   off += (size_t)out_size * sizeof(int2);

  k_convhat<<<F1 * NN / 256, 256, 0, stream>>>(x_hat, psi_r, psi_i, cplx);
  k_psi2sw <<<NCHUNK * 96 / 256, 256, 0, stream>>>(psi_r2, psi_i2, psi2sw);
  fft1d    <<<F1 * NSIDE, 128, 0, stream>>>(cplx, 0, +1.0f, 1.0f / 256.0f);  // ifft rows
  fft1d    <<<F1 * NSIDE, 128, 0, stream>>>(cplx, 1, +1.0f, 1.0f / 256.0f);  // ifft cols
  k_mod    <<<F1 * NN / 256, 256, 0, stream>>>(cplx);
  fft1d    <<<F1 * NSIDE, 128, 0, stream>>>(cplx, 0, -1.0f, 1.0f);           // fft rows
  fft1d    <<<F1 * NSIDE, 128, 0, stream>>>(cplx, 1, -1.0f, 1.0f);           // fft cols

  dim3 g(NTILE, KSEG);
  k_gemm    <<<g, 256, 0, stream>>>(cplx, (const v16bf*)psi2sw, partial);
  k_finalize<<<NPAIR_PAD * F2P / 256, 256, 0, stream>>>(partial, mags);
  k_buildmap<<<1, 1, 0, stream>>>(map);
  k_gather  <<<(out_size + 255) / 256, 256, 0, stream>>>(map, mags, (float*)d_out, out_size);
}
